// myConv2dCpu_62904091017749
// MI455X (gfx1250) — matrix operand order repaired and checked by reading, not compiled, not run on hardware
//
#include <hip/hip_runtime.h>

// ---------------------------------------------------------------------------
// Implicit-GEMM 3x3 VALID conv (NCHW/OIHW, fp32 in/out) for gfx1250 / MI455X.
// fp32 via bf16 hi/lo split: acc += Ahi*Bhi + Ahi*Blo + Alo*Bhi
// with v_wmma_f32_16x16x32_bf16 (f32 accumulate).
// GEMM: M=256 (COUT), N=123008 (B*62*62), K=1152.
// K is PERMUTED as k' = (kh*3+kw)*128 + ci (applied identically to A and B),
// so each 32-wide K-step has a fixed filter tap and contiguous ci:
//   - B im2col gather is affine (base + j*4096): no per-element div/mod
//   - A weight gather is affine (base + j*9)
// Block tile 128x128, K-step 32, 8 waves (wave32), each wave 64x32.
// Double-buffered LDS (1 barrier/iter), register prefetch of next K-step,
// non-temporal output stores to keep the 67MB input resident in 192MB L2.
// ---------------------------------------------------------------------------

typedef __attribute__((ext_vector_type(16))) __bf16 v16bf;
typedef __attribute__((ext_vector_type(8)))  __bf16 v8bf;
typedef __attribute__((ext_vector_type(8)))  float  v8f;

#define B_    32
#define CIN_  128
#define H_    64
#define W_    64
#define COUT_ 256
#define OH_   62
#define OW_   62
#define KDIM  (CIN_ * 9)        // 1152
#define HW_   (OH_ * OW_)       // 3844
#define NTOT  (B_ * HW_)        // 123008
#define INB   (CIN_ * H_ * W_)  // 524288 floats per batch image
#define OUTB  (COUT_ * HW_)     // 984064 floats per batch image
#define CHW_  (H_ * W_)         // 4096

#define MT 128
#define NT 128
#define KT 32
#define KSTEPS (KDIM / KT)      // 36 = 9 taps x 4 ci-blocks

__device__ __forceinline__ unsigned short f2bf_rne(float f) {
    unsigned int u = __float_as_uint(f);
    u += 0x7FFFu + ((u >> 16) & 1u);
    return (unsigned short)(u >> 16);
}
__device__ __forceinline__ float bf2f(unsigned short s) {
    return __uint_as_float(((unsigned int)s) << 16);
}

union Frag16 { v16bf v; v8bf h[2]; };

__global__ __launch_bounds__(256, 1)
void conv3x3_igemm_bf16x3(const float* __restrict__ x,
                          const float* __restrict__ wgt,
                          const float* __restrict__ bias,
                          float* __restrict__ out)
{
    // Double-buffered LDS tiles, bf16 hi/lo planes, K contiguous per row.
    // 2 * (8+8+8+8) KB = 64 KB of the 320 KB WGP pool.
    __shared__ __attribute__((aligned(64))) unsigned short Ah[2][MT * KT];
    __shared__ __attribute__((aligned(64))) unsigned short Al[2][MT * KT];
    __shared__ __attribute__((aligned(64))) unsigned short Bh[2][NT * KT];
    __shared__ __attribute__((aligned(64))) unsigned short Bl[2][NT * KT];

    const int tid   = threadIdx.x;
    const int lane  = tid & 31;     // wave32
    const int wv    = tid >> 5;     // 0..7
    const int waveM = wv >> 2;      // 0..1  (64 rows of M each)
    const int waveN = wv & 3;       // 0..3  (32 cols of N each)

    const int m0 = blockIdx.y * MT;
    const int n0 = blockIdx.x * NT;

    // ---- A-tile loader mapping: thread covers half a row (16 of 32 k') ---
    const int a_m = tid >> 1;              // 0..127
    const int a_c = (tid & 1) * 16;        // ci sub-offset: 0 or 16
    const float* a_row = wgt + (long)(m0 + a_m) * KDIM;

    // ---- B-tile loader mapping (im2col gather), n fixed per thread -------
    const int b_n    = tid & 127;          // 0..127
    const int b_c    = (tid >> 7) * 16;    // ci sub-offset: 0 or 16
    const int n_g    = n0 + b_n;
    const int bb     = n_g / HW_;
    const int hw     = n_g - bb * HW_;
    const int oh     = hw / OW_;
    const int ow     = hw - oh * OW_;
    const float* b_src = x + (long)bb * INB + oh * W_ + ow;

    v8f acc[4][2];
    #pragma unroll
    for (int mt = 0; mt < 4; ++mt)
        #pragma unroll
        for (int nt = 0; nt < 2; ++nt)
            acc[mt][nt] = (v8f)0.0f;

    const int lr = lane & 15;   // row/col within a 16x16 tile
    const int lh = lane >> 4;   // half-wave select

    // ---- register prefetch buffers + initial fill (K-step 0: tap 0) ------
    float aPre[16];
    float bPre[16];
    {
        // ks = 0 -> r9 = 0 (kh=kw=0), c0 = 0
        const float* ap = a_row + (long)(a_c) * 9;       // + r9(=0)
        const float* bp = b_src + (long)(b_c) * CHW_;    // + kh*64+kw(=0)
        #pragma unroll
        for (int j = 0; j < 16; ++j) aPre[j] = ap[j * 9];
        #pragma unroll
        for (int j = 0; j < 16; ++j) bPre[j] = bp[j * CHW_];
    }

    for (int ks = 0; ks < KSTEPS; ++ks) {
        const int buf = ks & 1;

        // ------- convert prefetched data, store into LDS buffer `buf` -----
        {
            alignas(16) unsigned short ht[16], lt[16];
            #pragma unroll
            for (int j = 0; j < 16; ++j) {
                const float f = aPre[j];
                const unsigned short h = f2bf_rne(f);
                ht[j] = h;
                lt[j] = f2bf_rne(f - bf2f(h));
            }
            unsigned short* dh = &Ah[buf][a_m * KT + a_c];
            unsigned short* dl = &Al[buf][a_m * KT + a_c];
            *(uint4*)(dh)     = *(const uint4*)&ht[0];
            *(uint4*)(dh + 8) = *(const uint4*)&ht[8];
            *(uint4*)(dl)     = *(const uint4*)&lt[0];
            *(uint4*)(dl + 8) = *(const uint4*)&lt[8];
        }
        {
            alignas(16) unsigned short ht[16], lt[16];
            #pragma unroll
            for (int j = 0; j < 16; ++j) {
                const float f = bPre[j];
                const unsigned short h = f2bf_rne(f);
                ht[j] = h;
                lt[j] = f2bf_rne(f - bf2f(h));
            }
            unsigned short* dh = &Bh[buf][b_n * KT + b_c];
            unsigned short* dl = &Bl[buf][b_n * KT + b_c];
            *(uint4*)(dh)     = *(const uint4*)&ht[0];
            *(uint4*)(dh + 8) = *(const uint4*)&ht[8];
            *(uint4*)(dl)     = *(const uint4*)&lt[0];
            *(uint4*)(dl + 8) = *(const uint4*)&lt[8];
        }

        __syncthreads();   // LDS buffer `buf` fully populated

        // ------- issue next K-step's global loads (overlap with WMMA) -----
        if (ks + 1 < KSTEPS) {
            const int ksn = ks + 1;
            const int r9  = ksn >> 2;            // filter tap 0..8 (uniform)
            const int kh  = r9 / 3;              // SALU magic-div, per step
            const int kw  = r9 - kh * 3;
            const int c0  = (ksn & 3) * 32;      // ci block base (uniform)

            const float* ap = a_row + (long)(c0 + a_c) * 9 + r9;
            const float* bp = b_src + (long)(c0 + b_c) * CHW_ + kh * W_ + kw;
            #pragma unroll
            for (int j = 0; j < 16; ++j) aPre[j] = ap[j * 9];
            #pragma unroll
            for (int j = 0; j < 16; ++j) bPre[j] = bp[j * CHW_];
        }

        // ------- fragments + WMMA from LDS buffer `buf` -------------------
        // B frag (32x16 bf16): lanes 0-15 -> K 0..15 at col n=lane%16,
        // lanes 16-31 -> K 16..31. Row-major [n][k] => one 32B read.
        Frag16 fbh[2], fbl[2];
        #pragma unroll
        for (int nt = 0; nt < 2; ++nt) {
            const int row = (waveN * 32 + nt * 16 + lr) * KT + lh * 16;
            const unsigned short* ph = &Bh[buf][row];
            const unsigned short* pl = &Bl[buf][row];
            fbh[nt].h[0] = *(const v8bf*)(ph);
            fbh[nt].h[1] = *(const v8bf*)(ph + 8);
            fbl[nt].h[0] = *(const v8bf*)(pl);
            fbl[nt].h[1] = *(const v8bf*)(pl + 8);
        }

        // A frag (16x32 bf16): lanes 0-15 hold K 0..15 of row m,
        // lanes 16-31 hold K 16..31. Two 16B reads per plane.
        #pragma unroll
        for (int mt = 0; mt < 4; ++mt) {
            const int row = (waveM * 64 + mt * 16 + lr) * KT + lh * 16;
            const unsigned short* ph = &Ah[buf][row];
            const unsigned short* pl = &Al[buf][row];
            Frag16 fah, fal;
            fah.h[0] = *(const v8bf*)(ph);
            fah.h[1] = *(const v8bf*)(ph + 8);
            fal.h[0] = *(const v8bf*)(pl);
            fal.h[1] = *(const v8bf*)(pl + 8);

            #pragma unroll
            for (int nt = 0; nt < 2; ++nt) {
                acc[mt][nt] = __builtin_amdgcn_wmma_f32_16x16x32_bf16(
                    false, fah.v, false, fbh[nt].v, (short)0, acc[mt][nt], false, false);
                acc[mt][nt] = __builtin_amdgcn_wmma_f32_16x16x32_bf16(
                    false, fah.v, false, fbl[nt].v, (short)0, acc[mt][nt], false, false);
                acc[mt][nt] = __builtin_amdgcn_wmma_f32_16x16x32_bf16(
                    false, fal.v, false, fbh[nt].v, (short)0, acc[mt][nt], false, false);
            }
        }
        // No trailing barrier: next iteration writes the *other* LDS buffer,
        // and passing the next barrier implies all waves finished reading
        // this one.
    }

    // ---------------- epilogue: bias + NCHW scatter ------------------------
    // D layout: VGPR r, lane l -> M = r + 8*(l>=16), N = l%16.
    // Output is write-once streaming: non-temporal stores so the 126MB
    // output doesn't evict the input working set from L2.
    #pragma unroll
    for (int mt = 0; mt < 4; ++mt) {
        const int mbase = m0 + waveM * 64 + mt * 16 + lh * 8;
        #pragma unroll
        for (int nt = 0; nt < 2; ++nt) {
            const int n_gl = n0 + waveN * 32 + nt * 16 + lr;
            const int bb2  = n_gl / HW_;
            const int hw2  = n_gl - bb2 * HW_;
            float* obase = out + (long)bb2 * OUTB + hw2;
            const v8f a = acc[mt][nt];
            #pragma unroll
            for (int r = 0; r < 8; ++r) {
                const int co = mbase + r;
                __builtin_nontemporal_store(a[r] + bias[co],
                                            &obase[(long)co * HW_]);
            }
        }
    }
}

extern "C" void kernel_launch(void* const* d_in, const int* in_sizes, int n_in,
                              void* d_out, int out_size, void* d_ws, size_t ws_size,
                              hipStream_t stream) {
    (void)in_sizes; (void)n_in; (void)out_size; (void)d_ws; (void)ws_size;
    const float* x    = (const float*)d_in[0];
    const float* wgt  = (const float*)d_in[1];
    const float* bias = (const float*)d_in[2];
    float* out        = (float*)d_out;

    dim3 grid(NTOT / NT, COUT_ / MT);  // (961, 2), all tiles exact
    conv3x3_igemm_bf16x3<<<grid, dim3(256), 0, stream>>>(x, wgt, bias, out);
}
